// NodeModel_1_19292993094373
// MI455X (gfx1250) — matrix operand order repaired and checked by reading, not compiled, not run on hardware
//
#include <hip/hip_runtime.h>
#include <hip/hip_bf16.h>

typedef __attribute__((ext_vector_type(16))) _Float16 v16h;
typedef __attribute__((ext_vector_type(8)))  float    v8f;

union F16x16 {
    v16h   v;
    _Float16 h[16];
    float4 q[2];
};

#define WAVES1 4   // waves per block, edge kernel
#define WAVES2 4   // waves per block, node kernel

// ---------------------------------------------------------------------------
// Fragment helpers.
//
// A-matrix (16x32 f16) per-lane layout used here (contiguous k, as B):
//   lane = hi*16 + m  (m = row 0..15, hi = 0/1)
//   halves[0..7]  = K = kchunk*32 + hi*16 + (0..7)
//   halves[8..15] = K = kchunk*32 + hi*16 + 8 + (0..7)
//
// B-matrix (32x16 f16) per-lane layout used here (contiguous k, as A):
//   lane = hi*16 + n  (n = col 0..15, hi = 0/1)
//   halves[i] = K = kchunk*32 + hi*16 + i
// We pre-swizzle weights so each lane's 16 halves are contiguous (32 bytes).
// ---------------------------------------------------------------------------

__device__ inline v16h load_a_frag(const _Float16* tile, int stride, int kchunk, int lane)
{
    int m  = lane & 15;
    int hi = lane >> 4;
    const _Float16* p = tile + m * stride + kchunk * 32 + hi * 16;
    F16x16 f;
    f.q[0] = *(const float4*)(p);
    f.q[1] = *(const float4*)(p + 8);
    return f.v;
}

__device__ inline v16h load_b_frag(const _Float16* fragbuf, int ntiles, int kchunk, int ntile, int lane)
{
    const _Float16* p = fragbuf + ((size_t)((kchunk * ntiles + ntile) * 32 + lane)) * 16;
    F16x16 f;
    f.q[0] = *(const float4*)(p);
    f.q[1] = *(const float4*)(p + 8);
    return f.v;
}

__device__ inline v8f wmma16(v16h a, v16h b, v8f c)
{
    return __builtin_amdgcn_wmma_f32_16x16x32_f16(false, a, false, b, (short)0, c, false, false);
}

// ---------------------------------------------------------------------------
// Prep: swizzle f32 row-major W[fan_in][fan_out] into f16 B-fragment layout.
// frag flat index f: i = f&15, lane = (f>>4)&31, fr = f>>9,
//                    ntile = fr % ntiles, kchunk = fr / ntiles
// ---------------------------------------------------------------------------
__global__ void prep_wfrag_kernel(const float* __restrict__ W, int fan_in, int fan_out,
                                  _Float16* __restrict__ frag, int kchunks, int ntiles)
{
    int total = kchunks * ntiles * 512;
    for (int f = blockIdx.x * blockDim.x + threadIdx.x; f < total;
         f += gridDim.x * blockDim.x) {
        int i      = f & 15;
        int lane   = (f >> 4) & 31;
        int fr     = f >> 9;
        int ntile  = fr % ntiles;
        int kchunk = fr / ntiles;
        int n = ntile * 16 + (lane & 15);
        int k = kchunk * 32 + ((lane >> 4) << 4) + i;
        float v = (k < fan_in && n < fan_out) ? W[(size_t)k * fan_out + n] : 0.0f;
        frag[f] = (_Float16)v;
    }
}

// ---------------------------------------------------------------------------
// Edge kernel: per-wave 16-edge tile. concat(x[col], edge_attr) -> 43 (pad 64)
// layer1: [16x64]x[64x64] (K=43 real), layer2: [16x64]x[64x128].
// Scatter f32 atomics into sum[N][128], cnt[N].
// ---------------------------------------------------------------------------
__global__ void __launch_bounds__(WAVES1 * 32)
edge_mlp_scatter_kernel(const float* __restrict__ x,
                        const long long* __restrict__ ei,   // [2][E]
                        const float* __restrict__ ea,       // [E][39]
                        const float* __restrict__ b1,
                        const float* __restrict__ b2,
                        const _Float16* __restrict__ w1f,
                        const _Float16* __restrict__ w2f,
                        float* __restrict__ sum,
                        float* __restrict__ cnt,
                        int E)
{
    __shared__ __align__(16) _Float16 feat[WAVES1][16][64];
    __shared__ __align__(16) _Float16 h1[WAVES1][16][64];
    __shared__ int rows[WAVES1][16];

    const int wave = threadIdx.x >> 5;
    const int lane = threadIdx.x & 31;
    const long long tile = (long long)blockIdx.x * WAVES1 + wave;
    const long long e0 = tile * 16;
    if (e0 >= E) return;                       // wave-uniform exit

    // ---- stage features into wave-private LDS (f16, K padded to 64) ----
    if (lane < 16) {
        long long e  = e0 + lane;
        bool valid   = e < E;
        long long ec = valid ? e : (long long)(E - 1);
        int col = (int)ei[(long long)E + ec];  // edge_index[1]
        int row = (int)ei[ec];                 // edge_index[0]
        rows[wave][lane] = valid ? row : -1;
        const float* xp = x + (long long)col * 4;
#pragma unroll
        for (int k = 0; k < 4; ++k)
            feat[wave][lane][k] = valid ? (_Float16)xp[k] : (_Float16)0.0f;
#pragma unroll
        for (int k = 43; k < 64; ++k)
            feat[wave][lane][k] = (_Float16)0.0f;
    }
    // edge_attr: coalesced sweep over 16*39 contiguous floats
    for (int idx = lane; idx < 16 * 39; idx += 32) {
        int r = idx / 39, c = idx - r * 39;
        long long e = e0 + r;
        float v = (e < E) ? ea[e * 39 + c] : 0.0f;
        feat[wave][r][4 + c] = (_Float16)v;
    }

    const int nl = lane & 15;
    const int hi = lane >> 4;

    // ---- layer 1: K = 2 chunks, N = 4 tiles ----
    v16h a0 = load_a_frag(&feat[wave][0][0], 64, 0, lane);
    v16h a1 = load_a_frag(&feat[wave][0][0], 64, 1, lane);
    v8f acc1[4];
#pragma unroll
    for (int nt = 0; nt < 4; ++nt) {
        float bb = b1[nt * 16 + nl];
        v8f c;
#pragma unroll
        for (int r = 0; r < 8; ++r) c[r] = bb;
        c = wmma16(a0, load_b_frag(w1f, 4, 0, nt, lane), c);
        c = wmma16(a1, load_b_frag(w1f, 4, 1, nt, lane), c);
        acc1[nt] = c;
    }
    // relu -> h1 LDS (A-layout source for layer 2)
#pragma unroll
    for (int nt = 0; nt < 4; ++nt)
#pragma unroll
        for (int r = 0; r < 8; ++r) {
            float v = acc1[nt][r];
            v = v > 0.0f ? v : 0.0f;
            h1[wave][r + hi * 8][nt * 16 + nl] = (_Float16)v;
        }

    // ---- layer 2: K = 2 chunks, N = 8 tiles ----
    v16h a20 = load_a_frag(&h1[wave][0][0], 64, 0, lane);
    v16h a21 = load_a_frag(&h1[wave][0][0], 64, 1, lane);
#pragma unroll
    for (int nt = 0; nt < 8; ++nt) {
        float bb = b2[nt * 16 + nl];
        v8f c;
#pragma unroll
        for (int r = 0; r < 8; ++r) c[r] = bb;
        c = wmma16(a20, load_b_frag(w2f, 8, 0, nt, lane), c);
        c = wmma16(a21, load_b_frag(w2f, 8, 1, nt, lane), c);
        // relu + scatter-add into L2-resident accumulator
#pragma unroll
        for (int r = 0; r < 8; ++r) {
            int m   = r + hi * 8;
            int row = rows[wave][m];
            if (row >= 0) {
                float v = c[r];
                v = v > 0.0f ? v : 0.0f;
                unsafeAtomicAdd(&sum[(long long)row * 128 + nt * 16 + nl], v);
            }
        }
    }
    if (lane < 16) {
        int row = rows[wave][lane];
        if (row >= 0) unsafeAtomicAdd(&cnt[row], 1.0f);
    }
}

// ---------------------------------------------------------------------------
// Node kernel: per-wave 16-node tile. concat(x, mean) -> 132 (pad 160).
// layer3: [16x160]x[160x256], layer4: [16x256]x[256x256]. ReLU, f32 out.
// ---------------------------------------------------------------------------
__global__ void __launch_bounds__(WAVES2 * 32)
node_mlp_kernel(const float* __restrict__ x,
                const float* __restrict__ sum,
                const float* __restrict__ cnt,
                const float* __restrict__ b3,
                const float* __restrict__ b4,
                const _Float16* __restrict__ w3f,
                const _Float16* __restrict__ w4f,
                float* __restrict__ out,
                int N)
{
    __shared__ __align__(16) _Float16 fin[WAVES2][16][160];
    __shared__ __align__(16) _Float16 h3[WAVES2][16][256];
    __shared__ float inv[WAVES2][16];

    const int wave = threadIdx.x >> 5;
    const int lane = threadIdx.x & 31;
    const long long tile = (long long)blockIdx.x * WAVES2 + wave;
    const int n0 = (int)(tile * 16);
    if (n0 >= N) return;                        // wave-uniform exit

    if (lane < 16) {
        int node  = n0 + lane;
        bool v    = node < N;
        float c   = v ? cnt[node] : 1.0f;
        inv[wave][lane] = 1.0f / fmaxf(c, 1.0f);
        const float* xp = x + (long long)(v ? node : 0) * 4;
#pragma unroll
        for (int k = 0; k < 4; ++k)
            fin[wave][lane][k] = v ? (_Float16)xp[k] : (_Float16)0.0f;
#pragma unroll
        for (int k = 132; k < 160; ++k)
            fin[wave][lane][k] = (_Float16)0.0f;
    }
    // mean features: coalesced sweep over 16*128 contiguous floats
    for (int idx = lane; idx < 16 * 128; idx += 32) {
        int r = idx >> 7, c = idx & 127;
        int node = n0 + r;
        float v = (node < N) ? sum[(long long)node * 128 + c] * inv[wave][r] : 0.0f;
        fin[wave][r][4 + c] = (_Float16)v;
    }

    const int nl = lane & 15;
    const int hi = lane >> 4;

    // ---- layer 3: K = 5 chunks, N = 16 tiles ----
    v16h a3[5];
#pragma unroll
    for (int kc = 0; kc < 5; ++kc)
        a3[kc] = load_a_frag(&fin[wave][0][0], 160, kc, lane);
#pragma unroll
    for (int nt = 0; nt < 16; ++nt) {
        float bb = b3[nt * 16 + nl];
        v8f c;
#pragma unroll
        for (int r = 0; r < 8; ++r) c[r] = bb;
#pragma unroll
        for (int kc = 0; kc < 5; ++kc)
            c = wmma16(a3[kc], load_b_frag(w3f, 16, kc, nt, lane), c);
#pragma unroll
        for (int r = 0; r < 8; ++r) {
            float v = c[r];
            v = v > 0.0f ? v : 0.0f;
            h3[wave][r + hi * 8][nt * 16 + nl] = (_Float16)v;
        }
    }

    // ---- layer 4: K = 8 chunks, N = 16 tiles ----
    v16h a4[8];
#pragma unroll
    for (int kc = 0; kc < 8; ++kc)
        a4[kc] = load_a_frag(&h3[wave][0][0], 256, kc, lane);
#pragma unroll
    for (int nt = 0; nt < 16; ++nt) {
        float bb = b4[nt * 16 + nl];
        v8f c;
#pragma unroll
        for (int r = 0; r < 8; ++r) c[r] = bb;
#pragma unroll
        for (int kc = 0; kc < 8; ++kc)
            c = wmma16(a4[kc], load_b_frag(w4f, 16, kc, nt, lane), c);
#pragma unroll
        for (int r = 0; r < 8; ++r) {
            int node = n0 + r + hi * 8;
            if (node < N) {
                float v = c[r];
                v = v > 0.0f ? v : 0.0f;
                out[(long long)node * 256 + nt * 16 + nl] = v;
            }
        }
    }
}

// ---------------------------------------------------------------------------
extern "C" void kernel_launch(void* const* d_in, const int* in_sizes, int n_in,
                              void* d_out, int out_size, void* d_ws, size_t ws_size,
                              hipStream_t stream)
{
    const float*     x  = (const float*)d_in[0];
    const long long* ei = (const long long*)d_in[1];
    const float*     ea = (const float*)d_in[2];
    const float* W1 = (const float*)d_in[3];
    const float* b1 = (const float*)d_in[4];
    const float* W2 = (const float*)d_in[5];
    const float* b2 = (const float*)d_in[6];
    const float* W3 = (const float*)d_in[7];
    const float* b3 = (const float*)d_in[8];
    const float* W4 = (const float*)d_in[9];
    const float* b4 = (const float*)d_in[10];
    float* out = (float*)d_out;

    const int N = in_sizes[0] / 4;   // 100000
    const int E = in_sizes[2] / 39;  // 1600000

    // workspace layout
    char*  ws   = (char*)d_ws;
    float* sum  = (float*)ws;                                    // N*128 f32
    float* cntb = (float*)(ws + (size_t)N * 128 * sizeof(float)); // N f32
    _Float16* w1f = (_Float16*)(ws + (size_t)N * 129 * sizeof(float));
    _Float16* w2f = w1f + 2 * 4  * 512;   //  4096 halves
    _Float16* w3f = w2f + 2 * 8  * 512;   //  8192 halves
    _Float16* w4f = w3f + 5 * 16 * 512;   // 40960 halves
    //          w4f + 8*16*512 = 65536 halves end

    // zero accumulators (sum and cnt are contiguous)
    hipMemsetAsync(sum, 0, (size_t)N * 129 * sizeof(float), stream);

    // pre-swizzle weights into f16 B-fragment layout
    prep_wfrag_kernel<<<16, 256, 0, stream>>>(W1, 43, 64, w1f, 2, 4);
    prep_wfrag_kernel<<<32, 256, 0, stream>>>(W2, 64, 128, w2f, 2, 8);
    prep_wfrag_kernel<<<160, 256, 0, stream>>>(W3, 132, 256, w3f, 5, 16);
    prep_wfrag_kernel<<<256, 256, 0, stream>>>(W4, 256, 256, w4f, 8, 16);

    // edge phase
    int tilesE  = (E + 15) / 16;
    int blocksE = (tilesE + WAVES1 - 1) / WAVES1;
    edge_mlp_scatter_kernel<<<blocksE, WAVES1 * 32, 0, stream>>>(
        x, ei, ea, b1, b2, w1f, w2f, sum, cntb, E);

    // node phase
    int tilesN  = (N + 15) / 16;
    int blocksN = (tilesN + WAVES2 - 1) / WAVES2;
    node_mlp_kernel<<<blocksN, WAVES2 * 32, 0, stream>>>(
        x, sum, cntb, b3, b4, w3f, w4f, out, N);
}
